// SememeSumLstm_81999515615468
// MI455X (gfx1250) — matrix operand order repaired and checked by reading, not compiled, not run on hardware
//
#include <hip/hip_runtime.h>
#include <hip/hip_bf16.h>

typedef __bf16 bf16_t;
typedef __attribute__((ext_vector_type(16))) __bf16 v16bf;
typedef __attribute__((ext_vector_type(4)))  __bf16 v4bf;
typedef __attribute__((ext_vector_type(8)))  float  v8f;
typedef __attribute__((ext_vector_type(4)))  float  v4f;

#define T_STEPS 512
#define BATCH   256
#define IN_DIM  300
#define MEM_DIM 300
#define NCOLS   1200            // 3*M (iou) + M (f)
#define NPAD    1216            // padded fused output width (76 n-tiles)
#define KT      20              // k-tiles: 0..9 x-part (K 0..319), 10..19 h-part (K 320..639)
#define KX      10

#define NBLOCKS  76
#define NTHREADS (NBLOCKS * 256)        // 19456
#define GATE_N   (BATCH * MEM_DIM)      // 76800
#define GATE_Q   (GATE_N / 4)           // 19200 quads (one per thread)

#define ABUF_ELEMS (16 * KT * 512)      // full A operand per step: 163840 bf16 (320 KB)
#define XFRAGS     (16 * KX * 32)       // x-part fragments per step: 5120

#define ABUF_OFF 0u                     // two parity buffers: 640 KB
#define Z_OFF    (1024u * 1024u)        // z: 256*1216 f32 = 1.19 MB
#define BAR_OFF  (2560u * 1024u)

// ---------------------------------------------------------------------------
// Device-wide sense-reversal barrier (agent scope); all blocks co-resident.
// ---------------------------------------------------------------------------
__device__ inline void grid_barrier(unsigned* cnt, unsigned* gen) {
  __threadfence();                 // release
  __syncthreads();
  if (threadIdx.x == 0) {
    unsigned g = __hip_atomic_load(gen, __ATOMIC_RELAXED, __HIP_MEMORY_SCOPE_AGENT);
    unsigned a = __hip_atomic_fetch_add(cnt, 1u, __ATOMIC_ACQ_REL, __HIP_MEMORY_SCOPE_AGENT);
    if (a == NBLOCKS - 1) {
      __hip_atomic_store(cnt, 0u, __ATOMIC_RELAXED, __HIP_MEMORY_SCOPE_AGENT);
      __hip_atomic_fetch_add(gen, 1u, __ATOMIC_RELEASE, __HIP_MEMORY_SCOPE_AGENT);
    } else {
      while (__hip_atomic_load(gen, __ATOMIC_ACQUIRE, __HIP_MEMORY_SCOPE_AGENT) == g)
        __builtin_amdgcn_s_sleep(2);
    }
  }
  __syncthreads();
  __threadfence();                 // acquire
}

__global__ void init_barrier(unsigned* bar) {
  if (threadIdx.x < 2) bar[threadIdx.x] = 0u;
}

// ---------------------------------------------------------------------------
// Pack one 16-element A fragment (lane ln of k-tile ktile, m-tile mtile) of
// x_t. A-frag map: lane l -> row m = l&15; elem j -> koff = (l<16?0:8) +
// (j<8 ? j : j+8)  => two contiguous 8-float runs at kb and kb+16.
// ---------------------------------------------------------------------------
__device__ inline void pack_x_frag(const float* __restrict__ xt,
                                   bf16_t* __restrict__ abuf, int f) {
  int mtile = f / (KX * 32);
  int rem   = f - mtile * (KX * 32);
  int ktile = rem >> 5;
  int ln    = rem & 31;
  int row   = mtile * 16 + (ln & 15);
  int kb    = ktile * 32 + ((ln < 16) ? 0 : 8);
  const float* p = xt + row * IN_DIM;
  v16bf av;
  if (ktile < KX - 1) {                  // fully in range (max k = 287 < 300)
    v4f c0 = *(const v4f*)(p + kb);
    v4f c1 = *(const v4f*)(p + kb + 4);
    v4f c2 = *(const v4f*)(p + kb + 16);
    v4f c3 = *(const v4f*)(p + kb + 20);
    #pragma unroll
    for (int e = 0; e < 4; ++e) {
      av[e]      = (bf16_t)c0[e];
      av[4 + e]  = (bf16_t)c1[e];
      av[8 + e]  = (bf16_t)c2[e];
      av[12 + e] = (bf16_t)c3[e];
    }
  } else {                               // last x k-tile: guard k < 300
    #pragma unroll
    for (int e = 0; e < 16; ++e) {
      int k = kb + ((e < 8) ? e : (e + 8));
      av[e] = (k < IN_DIM) ? (bf16_t)p[k] : (bf16_t)0.0f;
    }
  }
  *(v16bf*)(abuf + (mtile * KT + ktile) * 512 + ln * 16) = av;
}

// ---------------------------------------------------------------------------
// Persistent LSTM scan. Block b owns output n-tile b (16 of 1216 fused cols).
// A operand (x_t | h_{t-1}) lives as bf16 fragments in a 2-deep ring in ws,
// produced cooperatively one step ahead; B (weights) lives in LDS.
// ---------------------------------------------------------------------------
__global__ __launch_bounds__(256, 1)
void lstm_persistent(const float* __restrict__ x,        // [T,B,IN_DIM]
                     const float* __restrict__ Wioux,    // [900, IN_DIM]
                     const float* __restrict__ Wfx,      // [300, IN_DIM]
                     const float* __restrict__ Wiouh,    // [900, MEM_DIM]
                     const float* __restrict__ Wfh,      // [300, MEM_DIM]
                     const float* __restrict__ b_ioux,
                     const float* __restrict__ b_iouh,
                     const float* __restrict__ b_fx,
                     const float* __restrict__ b_fh,
                     bf16_t*      __restrict__ abuf,     // [2][ABUF_ELEMS]
                     float*       __restrict__ zbuf,     // [BATCH, NPAD]
                     float*       __restrict__ c_out,    // [T,B,M]
                     float*       __restrict__ h_out,    // [T,B,M]
                     unsigned*    __restrict__ bar)      // {cnt, gen}
{
  __shared__ bf16_t wlds[KT * 512];     // this n-tile's weights, 20 KB, frag-major

  const int ntile = blockIdx.x;         // 0..75
  const int lane  = threadIdx.x & 31;
  const int wave  = threadIdx.x >> 5;   // 0..7 -> m-tiles {2w, 2w+1}
  const int lrow  = lane & 15;
  const int hi    = (lane >> 4) & 1;
  const int mt0   = wave * 2;
  const int mt1   = wave * 2 + 1;
  const int tid_g = blockIdx.x * 256 + threadIdx.x;

  // ---- Prologue 1: pack this n-tile's weights (f32 -> bf16 fragments) ----
  // B-frag map: lane l -> col n = l&15; elem j -> koff = (l<16?0:8)+(j<8?j:j+8)
  for (int i = threadIdx.x; i < KT * 512; i += 256) {
    int j  = i & 15;
    int ln = (i >> 4) & 31;
    int kt = i >> 9;
    int n    = ntile * 16 + (ln & 15);
    int koff = ((ln < 16) ? 0 : 8) + ((j < 8) ? j : (j + 8));
    float v = 0.0f;
    if (n < NCOLS) {
      if (kt < KX) {
        int k = kt * 32 + koff;
        if (k < IN_DIM)
          v = (n < 3 * MEM_DIM) ? Wioux[n * IN_DIM + k]
                                : Wfx[(n - 3 * MEM_DIM) * IN_DIM + k];
      } else {
        int k = (kt - KX) * 32 + koff;
        if (k < MEM_DIM)
          v = (n < 3 * MEM_DIM) ? Wiouh[n * MEM_DIM + k]
                                : Wfh[(n - 3 * MEM_DIM) * MEM_DIM + k];
      }
    }
    wlds[i] = (bf16_t)v;
  }

  // ---- Prologue 2: zero both A ring buffers (h_0 = 0 and K padding) ----
  {
    v16bf zv;
    #pragma unroll
    for (int e = 0; e < 16; ++e) zv[e] = (bf16_t)0.0f;
    for (int i = tid_g * 16; i < 2 * ABUF_ELEMS; i += NTHREADS * 16)
      *(v16bf*)(abuf + i) = zv;
  }

  // ---- Prologue 3: per-thread gate quad (4 consecutive m in one row) ----
  const bool gact = (tid_g < GATE_Q);
  int   gb = 0, gm = 0, hpq = 0;
  v4f   bi4 = {0,0,0,0}, bo4 = {0,0,0,0}, bu4 = {0,0,0,0}, bf4 = {0,0,0,0};
  float creg[4] = {0.f, 0.f, 0.f, 0.f};
  if (gact) {
    int e4 = tid_g * 4;
    gb = e4 / MEM_DIM;
    gm = e4 - gb * MEM_DIM;               // multiple of 4; quad stays in row
    bi4 = *(const v4f*)(b_ioux + gm)               + *(const v4f*)(b_iouh + gm);
    bo4 = *(const v4f*)(b_ioux + MEM_DIM + gm)     + *(const v4f*)(b_iouh + MEM_DIM + gm);
    bu4 = *(const v4f*)(b_ioux + 2 * MEM_DIM + gm) + *(const v4f*)(b_iouh + 2 * MEM_DIM + gm);
    bf4 = *(const v4f*)(b_fx + gm)                 + *(const v4f*)(b_fh + gm);
    int k    = 320 + gm;                  // h occupies K in [320,640)
    int ktf  = k >> 5;                    // 10..19
    int koff = k & 31;                    // multiple of 4 -> quad contiguous
    int grp  = (koff >> 3) & 1;
    int j    = (koff & 7) + ((koff & 16) ? 8 : 0);
    hpq = ((gb >> 4) * KT + ktf) * 512 + ((gb & 15) + grp * 16) * 16 + j;
  }

  // ---- Prologue 4: pack x_0 into parity-0 buffer ----
  __syncthreads();
  if (tid_g < XFRAGS) pack_x_frag(x, abuf, tid_g);

  grid_barrier(bar, bar + 1);             // weights + A(0) visible

  const v8f zero8 = {0.f, 0.f, 0.f, 0.f, 0.f, 0.f, 0.f, 0.f};

  for (int t = 0; t < T_STEPS; ++t) {
    // ================= GEMM phase =================
    // All loads are base + compile-time immediate offset (<= 19456 B).
    const bf16_t* __restrict__ ab  = abuf + (size_t)(t & 1) * ABUF_ELEMS;
    const bf16_t* __restrict__ pa0 = ab + (mt0 * KT) * 512 + lane * 16;
    const bf16_t* __restrict__ pa1 = ab + (mt1 * KT) * 512 + lane * 16;
    const bf16_t* __restrict__ pb  = wlds + lane * 16;

    // 4 independent accumulation chains (even/odd k-tiles) for WMMA ILP
    v8f acc0a = zero8, acc0b = zero8, acc1a = zero8, acc1b = zero8;

    #pragma unroll
    for (int kt = 0; kt < KT; kt += 2) {
      v16bf b0  = *(const v16bf*)(pb  + kt * 512);
      v16bf b1  = *(const v16bf*)(pb  + (kt + 1) * 512);
      v16bf a00 = *(const v16bf*)(pa0 + kt * 512);
      v16bf a10 = *(const v16bf*)(pa1 + kt * 512);
      v16bf a01 = *(const v16bf*)(pa0 + (kt + 1) * 512);
      v16bf a11 = *(const v16bf*)(pa1 + (kt + 1) * 512);
      acc0a = __builtin_amdgcn_wmma_f32_16x16x32_bf16(false, a00, false, b0,
                                                      (short)0, acc0a, false, false);
      acc1a = __builtin_amdgcn_wmma_f32_16x16x32_bf16(false, a10, false, b0,
                                                      (short)0, acc1a, false, false);
      acc0b = __builtin_amdgcn_wmma_f32_16x16x32_bf16(false, a01, false, b1,
                                                      (short)0, acc0b, false, false);
      acc1b = __builtin_amdgcn_wmma_f32_16x16x32_bf16(false, a11, false, b1,
                                                      (short)0, acc1b, false, false);
    }
    v8f acc0 = acc0a + acc0b;
    v8f acc1 = acc1a + acc1b;

    // scatter z: lane l, vgpr r -> row = r + (l<16?0:8), col = l&15
    {
      const int col = ntile * 16 + lrow;
      #pragma unroll
      for (int r = 0; r < 8; ++r) {
        zbuf[(mt0 * 16 + r + hi * 8) * NPAD + col] = acc0[r];
        zbuf[(mt1 * 16 + r + hi * 8) * NPAD + col] = acc1[r];
      }
    }

    grid_barrier(bar, bar + 1);           // z visible to everyone

    // ================= Gate phase (+ pack next A buffer) =================
    bf16_t* __restrict__ abn = abuf + (size_t)((t + 1) & 1) * ABUF_ELEMS;

    if (gact) {
      const float* zr = zbuf + gb * NPAD;
      v4f zi = *(const v4f*)(zr + gm)               + bi4;
      v4f zo = *(const v4f*)(zr + MEM_DIM + gm)     + bo4;
      v4f zu = *(const v4f*)(zr + 2 * MEM_DIM + gm) + bu4;
      v4f zf = *(const v4f*)(zr + 3 * MEM_DIM + gm) + bf4;
      v4f cv, hv;
      v4bf hb;
      #pragma unroll
      for (int e = 0; e < 4; ++e) {
        float ig = 1.0f / (1.0f + __expf(-zi[e]));
        float og = 1.0f / (1.0f + __expf(-zo[e]));
        float ug = tanhf(zu[e]);
        float fg = 1.0f / (1.0f + __expf(-zf[e]));
        float c = ig * ug + fg * creg[e];
        creg[e] = c;
        float h = og * tanhf(c);
        cv[e] = c;
        hv[e] = h;
        hb[e] = (bf16_t)h;
      }
      size_t o = (size_t)t * GATE_N + (size_t)tid_g * 4;
      *(v4f*)(c_out + o) = cv;
      *(v4f*)(h_out + o) = hv;
      *(v4bf*)(abn + hpq) = hb;           // 4 contiguous bf16: one 8B store
    }

    // pack x_{t+1} fragments (done once, cooperatively, off the GEMM path)
    if (t + 1 < T_STEPS) {
      if (tid_g < XFRAGS)
        pack_x_frag(x + (size_t)(t + 1) * BATCH * IN_DIM, abn, tid_g);
      // prefetch x_{t+2} (global_prefetch_b8)
      if (t + 2 < T_STEPS) {
        int off = tid_g * 4;
        if (off < BATCH * IN_DIM)
          __builtin_prefetch(x + (size_t)(t + 2) * BATCH * IN_DIM + off, 0, 1);
      }
    }

    grid_barrier(bar, bar + 1);           // next A buffer visible before next GEMM
  }
}

extern "C" void kernel_launch(void* const* d_in, const int* in_sizes, int n_in,
                              void* d_out, int out_size, void* d_ws, size_t ws_size,
                              hipStream_t stream) {
  (void)in_sizes; (void)n_in; (void)out_size; (void)ws_size;

  const float* x     = (const float*)d_in[0];
  const float* Wioux = (const float*)d_in[1];
  const float* bioux = (const float*)d_in[2];
  const float* Wiouh = (const float*)d_in[3];
  const float* biouh = (const float*)d_in[4];
  const float* Wfx   = (const float*)d_in[5];
  const float* bfx   = (const float*)d_in[6];
  const float* Wfh   = (const float*)d_in[7];
  const float* bfh   = (const float*)d_in[8];

  char*     ws   = (char*)d_ws;
  bf16_t*   abuf = (bf16_t*)(ws + ABUF_OFF);
  float*    zbuf = (float*)(ws + Z_OFF);
  unsigned* bar  = (unsigned*)(ws + BAR_OFF);

  float* c_out = (float*)d_out;
  float* h_out = c_out + (size_t)T_STEPS * BATCH * MEM_DIM;

  init_barrier<<<1, 64, 0, stream>>>(bar);
  lstm_persistent<<<NBLOCKS, 256, 0, stream>>>(
      x, Wioux, Wfx, Wiouh, Wfh, bioux, biouh, bfx, bfh,
      abuf, zbuf, c_out, h_out, bar);
}
